// LowRankRNN_32667521254229
// MI455X (gfx1250) — 8-bit matrix operand order repaired and checked by reading, compiled, not run on hardware
//
#include <hip/hip_runtime.h>
#include <stdint.h>


#define NDIM        2048
#define TSTEPS      8192
#define NWG         16
#define ROWS_PER_WG 128
#define ROWSTRIDE   2064
#define SLAB_BYTES  (ROWS_PER_WG * ROWSTRIDE)
#define X8_OFF      SLAB_BYTES
#define STG_OFF     (SLAB_BYTES + 2048)
#define H_OFF       (SLAB_BYTES + 2048 + 512)
#define DYN_LDS     (SLAB_BYTES + 2048 + 512 + 512)

#define GG    2.0f
#define BB    10.0f
#define MM    1.5f
#define COEF  0.001f
#define SJ    1024.0f
#define SX    4.0f
#define INV_S (1.0f / (SJ * SX))

#define J8_OFF   0
#define J8_BYTES ((size_t)NWG * SLAB_BYTES)
#define XPUB_OFF J8_BYTES
#define CNT_OFF  (XPUB_OFF + (size_t)2 * NDIM * 4)

typedef int   v16i __attribute__((ext_vector_type(16)));
typedef float v8f  __attribute__((ext_vector_type(8)));

__device__ __forceinline__ unsigned char f32_to_e4m3(float x) {
  union { float f; unsigned u; } v;
  v.f = x;
  unsigned sgn = (v.u >> 31) << 7;
  unsigned au  = v.u & 0x7FFFFFFFu;
  v.u = au;
  float a = v.f;
  if (a >= 448.0f)          return (unsigned char)(sgn | 0x7Eu);
  if (a < 0.0009765625f)    return (unsigned char)sgn;
  int e = (int)(au >> 23) - 127;
  if (e >= -6) {
    unsigned lsb = (au >> 20) & 1u;
    unsigned r   = au + 0x0007FFFFu + lsb;
    e = (int)(r >> 23) - 127;
    unsigned m3 = (r >> 20) & 7u;
    if (e > 8 || (e == 8 && m3 == 7u)) return (unsigned char)(sgn | 0x7Eu);
    return (unsigned char)(sgn | ((unsigned)(e + 7) << 3) | m3);
  }
  int m = (int)(a * 512.0f + 0.5f);
  if (m >= 8) return (unsigned char)(sgn | 0x08u);
  if (m <= 0) return (unsigned char)sgn;
  return (unsigned char)(sgn | (unsigned)m);
}

__device__ __forceinline__ v16i load_a_frag(const char* aBase, int kt) {
  uint4 c0 = *(const uint4*)(aBase + kt);
  uint4 c1 = *(const uint4*)(aBase + kt + 32);
  uint4 c2 = *(const uint4*)(aBase + kt + 64);
  uint4 c3 = *(const uint4*)(aBase + kt + 96);
  v16i a;
  a[0]=(int)c0.x;  a[1]=(int)c0.y;  a[2]=(int)c0.z;  a[3]=(int)c0.w;
  a[4]=(int)c1.x;  a[5]=(int)c1.y;  a[6]=(int)c1.z;  a[7]=(int)c1.w;
  a[8]=(int)c2.x;  a[9]=(int)c2.y;  a[10]=(int)c2.z; a[11]=(int)c2.w;
  a[12]=(int)c3.x; a[13]=(int)c3.y; a[14]=(int)c3.z; a[15]=(int)c3.w;
  return a;
}

__device__ __forceinline__ v16i load_b_frag(const unsigned char* xBase, int kt) {
  uint4 b0 = *(const uint4*)(xBase + kt);
  uint4 b1 = *(const uint4*)(xBase + kt + 32);
  uint4 b2 = *(const uint4*)(xBase + kt + 64);
  uint4 b3 = *(const uint4*)(xBase + kt + 96);
  v16i b;
  b[0]=(int)b0.x;  b[1]=(int)b0.y;  b[2]=(int)b0.z;  b[3]=(int)b0.w;
  b[4]=(int)b1.x;  b[5]=(int)b1.y;  b[6]=(int)b1.z;  b[7]=(int)b1.w;
  b[8]=(int)b2.x;  b[9]=(int)b2.y;  b[10]=(int)b2.z; b[11]=(int)b2.w;
  b[12]=(int)b3.x; b[13]=(int)b3.y; b[14]=(int)b3.z; b[15]=(int)b3.w;
  return b;
}

__global__ void prep_kernel(const float* __restrict__ W, const float* __restrict__ u,
                            const float* __restrict__ v, const float* __restrict__ h0,
                            unsigned char* __restrict__ j8, float* __restrict__ xpub,
                            unsigned* __restrict__ cnt) {
  const int row = blockIdx.x;
  const int wg  = row >> 7;
  const int r   = row & 127;
  unsigned char* dst = j8 + (size_t)wg * SLAB_BYTES + (size_t)r * ROWSTRIDE;
  const float* wrow = W + (size_t)row * NDIM;
  const float  um   = u[row] * MM;
  const float  off  = -BB / (float)NDIM;
  for (int c = threadIdx.x; c < NDIM; c += blockDim.x) {
    float jv = GG * wrow[c] + off + um * v[c];
    dst[c] = f32_to_e4m3(jv * SJ);
  }
  if (blockIdx.x < 8) {
    int i = blockIdx.x * 256 + threadIdx.x;
    xpub[i] = fmaxf(h0[i], 0.0f);
  }
  if (blockIdx.x == 0 && threadIdx.x == 0) *cnt = 0u;
}

__global__ __launch_bounds__(256, 1) void rnn_kernel(
    const float* __restrict__ I_t, const float* __restrict__ h0,
    const unsigned char* __restrict__ j8, float* __restrict__ xpub,
    unsigned* __restrict__ cnt, float* __restrict__ h_out) {
  extern __shared__ char smem[];
  const int tid = threadIdx.x;
  const int wg  = blockIdx.x;

  {
    const unsigned long long gbase =
        (unsigned long long)(uintptr_t)(j8 + (size_t)wg * SLAB_BYTES);
    const unsigned ldsbase = (unsigned)(uintptr_t)smem;
    for (int ofs = tid * 16; ofs < SLAB_BYTES; ofs += 256 * 16) {
      unsigned laddr = ldsbase + (unsigned)ofs;
      int      voff  = ofs;
      asm volatile("global_load_async_to_lds_b128 %0, %1, %2"
                   :: "v"(laddr), "v"(voff), "s"(gbase) : "memory");
    }
    asm volatile("s_wait_asynccnt 0" ::: "memory");
  }

  float*         lds_h = (float*)(smem + H_OFF);
  float*         stg   = (float*)(smem + STG_OFF);
  unsigned char* x8    = (unsigned char*)(smem + X8_OFF);
  if (tid < ROWS_PER_WG) lds_h[tid] = h0[wg * ROWS_PER_WG + tid];
  __syncthreads();

  const int lane = tid & 31;
  const int wave = tid >> 5;
  const int half = lane >> 4;
  const int m    = lane & 15;
  const char*          aBase = smem + (wave * 16 + m) * ROWSTRIDE + half * 16;
  const unsigned char* xBase = x8 + half * 16;

  for (int t = 0; t < TSTEPS; ++t) {
    const float* xcur = xpub + (t & 1) * NDIM;
    float*       xnxt = xpub + ((t + 1) & 1) * NDIM;

    for (int i = tid; i < NDIM; i += 256) {
      float xv = __hip_atomic_load(xcur + i, __ATOMIC_RELAXED, __HIP_MEMORY_SCOPE_AGENT);
      x8[i] = f32_to_e4m3(xv * SX);
    }
    __syncthreads();

    v8f acc0 = {0.f,0.f,0.f,0.f,0.f,0.f,0.f,0.f};
    v8f acc1 = {0.f,0.f,0.f,0.f,0.f,0.f,0.f,0.f};
#pragma unroll 2
    for (int kt = 0; kt < NDIM; kt += 256) {
      {
        v16i a = load_a_frag(aBase, kt);
        v16i b = load_b_frag(xBase, kt);
        acc0 = __builtin_amdgcn_wmma_f32_16x16x128_fp8_fp8(a, b, (short)0, acc0, false, false);
      }
      {
        v16i a = load_a_frag(aBase, kt + 128);
        v16i b = load_b_frag(xBase, kt + 128);
        acc1 = __builtin_amdgcn_wmma_f32_16x16x128_fp8_fp8(a, b, (short)0, acc1, false, false);
      }
    }

    if ((lane & 15) == 0) {
#pragma unroll
      for (int r = 0; r < 8; ++r)
        stg[wave * 16 + half * 8 + r] = (acc0[r] + acc1[r]) * INV_S;
    }
    __syncthreads();

    if (tid < ROWS_PER_WG) {
      const int g  = wg * ROWS_PER_WG + tid;
      float h  = lds_h[tid];
      float jx = stg[tid];
      float it = I_t[(size_t)t * NDIM + g];
      float hn = h + COEF * (jx + it - h);
      lds_h[tid] = hn;
      h_out[(size_t)t * NDIM + g] = hn;
      __hip_atomic_store(xnxt + g, fmaxf(hn, 0.0f),
                         __ATOMIC_RELAXED, __HIP_MEMORY_SCOPE_AGENT);
    }

    __syncthreads();
    __builtin_amdgcn_s_cluster_barrier();
    if (tid == 0) {
      __hip_atomic_fetch_add(cnt, 1u, __ATOMIC_ACQ_REL, __HIP_MEMORY_SCOPE_AGENT);
      const unsigned target = (unsigned)(t + 1) * NWG;
      while (__hip_atomic_load(cnt, __ATOMIC_ACQUIRE, __HIP_MEMORY_SCOPE_AGENT) < target)
        __builtin_amdgcn_s_sleep(1);
    }
    __syncthreads();
  }
}

__global__ void readout_kernel(const float* __restrict__ h_t, const float* __restrict__ ro_w,
                               const float* __restrict__ ro_b, float* __restrict__ y) {
  __shared__ float wsum[8];
  const int t = blockIdx.x;
  const float* h = h_t + (size_t)t * NDIM;
  float s = 0.f;
  for (int i = threadIdx.x; i < NDIM; i += 256) s += h[i] * ro_w[i];
  for (int o = 16; o > 0; o >>= 1) s += __shfl_xor(s, o, 32);
  if ((threadIdx.x & 31) == 0) wsum[threadIdx.x >> 5] = s;
  __syncthreads();
  if (threadIdx.x == 0) {
    float tot = 0.f;
#pragma unroll
    for (int w = 0; w < 8; ++w) tot += wsum[w];
    y[t] = tot + ro_b[0];
  }
}

extern "C" void kernel_launch(void* const* d_in, const int* in_sizes, int n_in,
                              void* d_out, int out_size, void* d_ws, size_t ws_size,
                              hipStream_t stream) {
  (void)in_sizes; (void)n_in; (void)out_size; (void)ws_size;
  const float* I_t  = (const float*)d_in[0];
  const float* h0   = (const float*)d_in[1];
  const float* W    = (const float*)d_in[2];
  const float* u    = (const float*)d_in[3];
  const float* v    = (const float*)d_in[4];
  const float* ro_w = (const float*)d_in[5];
  const float* ro_b = (const float*)d_in[6];

  float* y   = (float*)d_out;
  float* h_t = (float*)d_out + TSTEPS;

  unsigned char* j8   = (unsigned char*)d_ws + J8_OFF;
  float*         xpub = (float*)((char*)d_ws + XPUB_OFF);
  unsigned*      cnt  = (unsigned*)((char*)d_ws + CNT_OFF);

  (void)hipFuncSetAttribute(reinterpret_cast<const void*>(rnn_kernel),
                            hipFuncAttributeMaxDynamicSharedMemorySize, DYN_LDS);

  prep_kernel<<<NDIM, 256, 0, stream>>>(W, u, v, h0, j8, xpub, cnt);
  rnn_kernel<<<NWG, 256, DYN_LDS, stream>>>(I_t, h0, j8, xpub, cnt, h_t);
  readout_kernel<<<TSTEPS, 256, 0, stream>>>(h_t, ro_w, ro_b, y);
}
